// MoEDetector_17557826306729
// MI455X (gfx1250) — matrix operand order repaired and checked by reading, not compiled, not run on hardware
//
#include <hip/hip_runtime.h>
#include <cstdint>

#define THREADS 256
#define BM 128
#define BN 128
#define BK 64

static const int Bsz = 8, Sseq = 1024, Hdim = 1536, Mtok = 8192;

typedef __attribute__((ext_vector_type(16))) __bf16 v16bf;
typedef __attribute__((ext_vector_type(8)))  __bf16 v8bf;
typedef __attribute__((ext_vector_type(8)))  float  v8f;

union BF16x16 { v16bf v; v8bf h[2]; };

__device__ __forceinline__ unsigned short f2bf(float f) {
  unsigned u = __float_as_uint(f);
  unsigned r = u + 0x7FFFu + ((u >> 16) & 1u);
  return (unsigned short)(r >> 16);
}

__device__ __forceinline__ float gelu_exact(float x) {
  return 0.5f * x * (1.0f + erff(x * 0.70710678118654752440f));
}

// gfx1250 async global->LDS copy (16B per lane), tracked by ASYNCcnt.
// LDS operand = low 32 bits of the flat shared address (flat LDS aperture
// carries the workgroup-relative LDS byte offset in addr[31:0]).
__device__ __forceinline__ void async_copy_b128(const unsigned short* gptr,
                                                const unsigned short* lptr) {
  unsigned lds = (unsigned)(size_t)lptr;
  asm volatile("global_load_async_to_lds_b128 %0, %1, off"
               :: "v"(lds), "v"(gptr)
               : "memory");
}

__device__ __forceinline__ void wait_async0() {
  asm volatile("s_wait_asynccnt 0" ::: "memory");
}

// ---------------------------------------------------------------------------
// Templated BF16 WMMA GEMM:  C[M,N] = A[M,K] (row-major bf16) x B[N,K]^T
// (B operand stored N-major, contiguous over K).
// MODE 0: store bf16 transposed per-batch: out[(b*N+col)*sLocal + t]
// MODE 1: relu -> bf16 row-major
// MODE 2: relu -> f32 row-major
// MODE 3: expert epilogue: v = gate[row]*gelu(acc+bias[col]);
//         init ? out = sel==e ? v : 0  :  (sel==e ? out += v : nop)
// strideB != 0 => batched B: Bp = B + (blockRow/sLocal)*strideB
// Pipeline: double-buffered LDS, tiles staged with async global->LDS loads
// overlapping the WMMA compute of the previous tile.
// ---------------------------------------------------------------------------
template <int MODE>
__global__ __launch_bounds__(THREADS)
void gemm_wmma(const unsigned short* __restrict__ A,
               const unsigned short* __restrict__ B,
               void* __restrict__ Out,
               int M, int N, int K,
               long strideB, int sLocal,
               const float* __restrict__ bias,
               const int* __restrict__ sel,
               const float* __restrict__ gate,
               int expertId, int initFlag) {
  __shared__ unsigned short sA[2][BM * BK];   // [buf][row][k]  2 x 16 KB
  __shared__ unsigned short sB[2][BN * BK];   // [buf][col][k]  2 x 16 KB

  const int tid  = threadIdx.x;
  const int lane = tid & 31;
  const int wave = tid >> 5;
  const int wm   = wave >> 2;   // 0..1  (M dir, 64 rows each)
  const int wn   = wave & 3;    // 0..3  (N dir, 32 cols each)
  const int hl   = lane >> 4;   // lane half
  const int lr   = lane & 15;

  const int blockM = blockIdx.y * BM;
  const int blockN = blockIdx.x * BN;

  const unsigned short* Bp = B;
  if (strideB) Bp += (long)(blockM / sLocal) * strideB;

  // per-thread chunk mapping: 4 chunks of 16B for each of A and B per tile
  const int crow = tid >> 1;            // rows 0..127 (2 threads per row)
  const int cseg = (tid & 1) * 32;      // element offset: halves of BK=64
  const unsigned short* gA = A  + (long)(blockM + crow) * K + cseg;
  const unsigned short* gB = Bp + (long)(blockN + crow) * K + cseg;

  v8f acc[4][2];
#pragma unroll
  for (int m = 0; m < 4; ++m)
#pragma unroll
    for (int n = 0; n < 2; ++n)
#pragma unroll
      for (int i = 0; i < 8; ++i) acc[m][n][i] = 0.0f;

  const int nTiles = K / BK;

  // stage tile 0 into buffer 0 (each thread: 4x16B for A, 4x16B for B)
  {
    const unsigned short* lA = &sA[0][crow * BK + cseg];
    const unsigned short* lB = &sB[0][crow * BK + cseg];
    async_copy_b128(gA,      lA);
    async_copy_b128(gA + 8,  lA + 8);
    async_copy_b128(gA + 16, lA + 16);
    async_copy_b128(gA + 24, lA + 24);
    async_copy_b128(gB,      lB);
    async_copy_b128(gB + 8,  lB + 8);
    async_copy_b128(gB + 16, lB + 16);
    async_copy_b128(gB + 24, lB + 24);
  }
  wait_async0();
  __syncthreads();

  for (int t = 0; t < nTiles; ++t) {
    const int cur = t & 1;
    const bool hasNext = (t + 1 < nTiles);
    if (hasNext) {
      // buffer cur^1 was last read in iteration t-1, barriered since
      long k0 = (long)(t + 1) * BK;
      const unsigned short* lA = &sA[cur ^ 1][crow * BK + cseg];
      const unsigned short* lB = &sB[cur ^ 1][crow * BK + cseg];
      async_copy_b128(gA + k0,      lA);
      async_copy_b128(gA + k0 + 8,  lA + 8);
      async_copy_b128(gA + k0 + 16, lA + 16);
      async_copy_b128(gA + k0 + 24, lA + 24);
      async_copy_b128(gB + k0,      lB);
      async_copy_b128(gB + k0 + 8,  lB + 8);
      async_copy_b128(gB + k0 + 16, lB + 16);
      async_copy_b128(gB + k0 + 24, lB + 24);
    }

#pragma unroll
    for (int ks = 0; ks < 2; ++ks) {      // 2 WMMA K-steps per LDS stage
      BF16x16 afr[4], bfr[2];
#pragma unroll
      for (int m = 0; m < 4; ++m) {
        int row = wm * 64 + m * 16 + lr;
        int off = row * BK + ks * 32 + hl * 8;      // K = hl*8 + 0..7
        afr[m].h[0] = *(const v8bf*)(&sA[cur][off]);
        afr[m].h[1] = *(const v8bf*)(&sA[cur][off + 16]); // K = 16+hl*8+0..7
      }
#pragma unroll
      for (int n = 0; n < 2; ++n) {
        int col = wn * 32 + n * 16 + lr;
        int off = col * BK + ks * 32 + hl * 8;
        bfr[n].h[0] = *(const v8bf*)(&sB[cur][off]);
        bfr[n].h[1] = *(const v8bf*)(&sB[cur][off + 16]);
      }
#pragma unroll
      for (int m = 0; m < 4; ++m)
#pragma unroll
        for (int n = 0; n < 2; ++n)
          acc[m][n] = __builtin_amdgcn_wmma_f32_16x16x32_bf16(
              false, afr[m].v, false, bfr[n].v, (short)0, acc[m][n], false, false);
    }

    if (hasNext) {
      wait_async0();     // next tile fully landed in LDS (own wave's ops)
      __syncthreads();   // all waves landed + done reading current buffer
    }
  }

  // Epilogue. C layout: VGPR i -> row = tile + hl*8 + i, col = tile + lr.
  const int baseM = blockM + wm * 64;
  const int baseN = blockN + wn * 32;
#pragma unroll
  for (int m = 0; m < 4; ++m) {
#pragma unroll
    for (int n = 0; n < 2; ++n) {
      int col = baseN + n * 16 + lr;
#pragma unroll
      for (int i = 0; i < 8; ++i) {
        int   row = baseM + m * 16 + hl * 8 + i;
        float v   = acc[m][n][i];
        if (MODE == 0) {
          int bb = row / sLocal;
          int t2 = row - bb * sLocal;
          ((unsigned short*)Out)[((long)bb * N + col) * sLocal + t2] = f2bf(v);
        } else if (MODE == 1) {
          ((unsigned short*)Out)[(long)row * N + col] = f2bf(fmaxf(v, 0.0f));
        } else if (MODE == 2) {
          ((float*)Out)[(long)row * N + col] = fmaxf(v, 0.0f);
        } else {
          float  x   = v + bias[col];
          float  val = gate[row] * gelu_exact(x);
          bool   ch  = (sel[row] == expertId);
          float* o   = (float*)Out + (long)row * N + col;
          if (initFlag)      *o = ch ? val : 0.0f;
          else if (ch)       *o += val;
        }
      }
    }
  }
}

// ---------------------------------------------------------------------------
// Small helper kernels
// ---------------------------------------------------------------------------
__global__ __launch_bounds__(THREADS)
void k_cvt(const float* __restrict__ in, unsigned short* __restrict__ out, long n) {
  long i = (long)blockIdx.x * THREADS + threadIdx.x;
  if (i < n) out[i] = f2bf(in[i]);
}

// w[K][N] f32 -> wt[N][K] bf16
__global__ __launch_bounds__(THREADS)
void k_cvt_wt(const float* __restrict__ w, unsigned short* __restrict__ wt, int K, int N) {
  long i = (long)blockIdx.x * THREADS + threadIdx.x;
  if (i < (long)K * N) {
    int k = (int)(i / N), n = (int)(i % N);
    wt[(long)n * K + k] = f2bf(w[i]);
  }
}

// row-normalize adjacency, write bf16
__global__ __launch_bounds__(THREADS)
void k_adjprep(const float* __restrict__ adj, unsigned short* __restrict__ adjn) {
  int r = blockIdx.x;                       // global row b*S + s
  const float* row = adj + (long)r * Sseq;
  __shared__ float red[THREADS];
  float s = 0.0f;
  for (int t = threadIdx.x; t < Sseq; t += THREADS) s += row[t];
  red[threadIdx.x] = s;
  __syncthreads();
  for (int st = THREADS / 2; st >= 1; st >>= 1) {
    if (threadIdx.x < st) red[threadIdx.x] += red[threadIdx.x + st];
    __syncthreads();
  }
  float inv = 1.0f / fmaxf(red[0], 1e-9f);
  for (int t = threadIdx.x; t < Sseq; t += THREADS)
    adjn[(long)r * Sseq + t] = f2bf(row[t] * inv);
}

// router: logits, length masking, softmax, per-group top-1 + normalized gates
__global__ __launch_bounds__(THREADS)
void k_router(const float* __restrict__ hs, const float* __restrict__ rw,
              const float* __restrict__ rb, const int* __restrict__ seqlen,
              int* __restrict__ selSyn, int* __restrict__ selLen, int* __restrict__ selSem,
              float* __restrict__ gSyn, float* __restrict__ gLen, float* __restrict__ gSem) {
  int m = blockIdx.x;
  const float* x = hs + (long)m * Hdim;
  float p[8];
#pragma unroll
  for (int e = 0; e < 8; ++e) p[e] = 0.0f;
  for (int h = threadIdx.x; h < Hdim; h += THREADS) {
    float xv = x[h];
    const float* w = rw + (long)h * 8;
#pragma unroll
    for (int e = 0; e < 8; ++e) p[e] += xv * w[e];
  }
  __shared__ float red[THREADS][8];
#pragma unroll
  for (int e = 0; e < 8; ++e) red[threadIdx.x][e] = p[e];
  __syncthreads();
  for (int st = THREADS / 2; st >= 1; st >>= 1) {
    if (threadIdx.x < st)
#pragma unroll
      for (int e = 0; e < 8; ++e) red[threadIdx.x][e] += red[threadIdx.x + st][e];
    __syncthreads();
  }
  if (threadIdx.x == 0) {
    float lg[8];
#pragma unroll
    for (int e = 0; e < 8; ++e) lg[e] = red[0][e] + rb[e];
    int  b       = m >> 10;
    bool isShort = (seqlen[b] <= 128);
    if (isShort) lg[4] = -1e9f; else lg[3] = -1e9f;
    float mx = lg[0];
#pragma unroll
    for (int e = 1; e < 8; ++e) mx = fmaxf(mx, lg[e]);
    float sum = 0.0f;
#pragma unroll
    for (int e = 0; e < 8; ++e) { lg[e] = expf(lg[e] - mx); sum += lg[e]; }
    float inv = 1.0f / sum;
#pragma unroll
    for (int e = 0; e < 8; ++e) lg[e] *= inv;
    int  si = 0; float sp = lg[0];
    if (lg[1] > sp) { sp = lg[1]; si = 1; }
    if (lg[2] > sp) { sp = lg[2]; si = 2; }
    int  li = (lg[4] > lg[3]) ? 1 : 0;
    float lp = fmaxf(lg[3], lg[4]);
    int  mi = 0; float mp = lg[5];
    if (lg[6] > mp) { mp = lg[6]; mi = 1; }
    if (lg[7] > mp) { mp = lg[7]; mi = 2; }
    float tot = sp + lp + mp;
    float it  = 1.0f / tot;
    selSyn[m] = si; selLen[m] = li; selSem[m] = mi;
    gSyn[m] = sp * it; gLen[m] = lp * it; gSem[m] = mp * it;
  }
}

// shared = LayerNorm(x2 + hs) -> bf16 row-major
__global__ __launch_bounds__(THREADS)
void k_ln(const float* __restrict__ x2, const float* __restrict__ hs,
          const float* __restrict__ g, const float* __restrict__ bb,
          unsigned short* __restrict__ out) {
  int m = blockIdx.x;
  __shared__ float buf[Hdim];
  __shared__ float red[THREADS];
  const float* xr = x2 + (long)m * Hdim;
  const float* hr = hs + (long)m * Hdim;
  float s = 0.0f;
  for (int h = threadIdx.x; h < Hdim; h += THREADS) {
    float y = xr[h] + hr[h];
    buf[h]  = y;
    s      += y;
  }
  red[threadIdx.x] = s;
  __syncthreads();
  for (int st = THREADS / 2; st >= 1; st >>= 1) {
    if (threadIdx.x < st) red[threadIdx.x] += red[threadIdx.x + st];
    __syncthreads();
  }
  float mu = red[0] * (1.0f / Hdim);
  __syncthreads();
  float vs = 0.0f;
  for (int h = threadIdx.x; h < Hdim; h += THREADS) {
    float d = buf[h] - mu;
    vs += d * d;
  }
  red[threadIdx.x] = vs;
  __syncthreads();
  for (int st = THREADS / 2; st >= 1; st >>= 1) {
    if (threadIdx.x < st) red[threadIdx.x] += red[threadIdx.x + st];
    __syncthreads();
  }
  float rstd = rsqrtf(red[0] * (1.0f / Hdim) + 1e-5f);
  for (int h = threadIdx.x; h < Hdim; h += THREADS)
    out[(long)m * Hdim + h] = f2bf((buf[h] - mu) * rstd * g[h] + bb[h]);
}

// out[m,0:2] = fused[m,:] @ cls_w + cls_b
__global__ __launch_bounds__(THREADS)
void k_cls(const float* __restrict__ fused, const float* __restrict__ cw,
           const float* __restrict__ cb, float* __restrict__ out) {
  int m = blockIdx.x;
  __shared__ float r0[THREADS], r1[THREADS];
  float c0 = 0.0f, c1 = 0.0f;
  const float* f = fused + (long)m * Hdim;
  for (int h = threadIdx.x; h < Hdim; h += THREADS) {
    float v = f[h];
    c0 += v * cw[2 * h];
    c1 += v * cw[2 * h + 1];
  }
  r0[threadIdx.x] = c0; r1[threadIdx.x] = c1;
  __syncthreads();
  for (int st = THREADS / 2; st >= 1; st >>= 1) {
    if (threadIdx.x < st) {
      r0[threadIdx.x] += r0[threadIdx.x + st];
      r1[threadIdx.x] += r1[threadIdx.x + st];
    }
    __syncthreads();
  }
  if (threadIdx.x == 0) {
    out[2 * m]     = r0[0] + cb[0];
    out[2 * m + 1] = r1[0] + cb[1];
  }
}

// ---------------------------------------------------------------------------
extern "C" void kernel_launch(void* const* d_in, const int* in_sizes, int n_in,
                              void* d_out, int out_size, void* d_ws, size_t ws_size,
                              hipStream_t stream) {
  (void)in_sizes; (void)n_in; (void)out_size; (void)ws_size;
  const float* hs       = (const float*)d_in[0];
  const float* adj      = (const float*)d_in[1];
  const int*   seqlen   = (const int*)d_in[2];
  const float* router_w = (const float*)d_in[3];
  const float* router_b = (const float*)d_in[4];
  const float* gcn1_w   = (const float*)d_in[5];
  const float* gcn2_w   = (const float*)d_in[6];
  const float* ln_g     = (const float*)d_in[7];
  const float* ln_b     = (const float*)d_in[8];
  const float* syn_w    = (const float*)d_in[9];
  const float* syn_b    = (const float*)d_in[10];
  const float* lenS_w   = (const float*)d_in[11];
  const float* lenS_b   = (const float*)d_in[12];
  const float* lenL_w   = (const float*)d_in[13];
  const float* lenL_b   = (const float*)d_in[14];
  const float* sem_w    = (const float*)d_in[15];
  const float* sem_b    = (const float*)d_in[16];
  const float* cls_w    = (const float*)d_in[17];
  const float* cls_b    = (const float*)d_in[18];
  float* out = (float*)d_out;

  char*  ws  = (char*)d_ws;
  size_t off = 0;
  auto alloc = [&](size_t bytes) -> char* {
    char* p = ws + off;
    off += (bytes + 255) & ~(size_t)255;
    return p;
  };

  const long HH = (long)Hdim * Hdim;
  unsigned short* hsb   = (unsigned short*)alloc((size_t)Mtok * Hdim * 2);
  unsigned short* wt[10];
  for (int i = 0; i < 10; ++i) wt[i] = (unsigned short*)alloc((size_t)HH * 2);
  unsigned short* adjnb = (unsigned short*)alloc((size_t)Mtok * Sseq * 2);
  unsigned short* suppT = (unsigned short*)alloc((size_t)Bsz * Hdim * Sseq * 2);
  unsigned short* x1b   = (unsigned short*)alloc((size_t)Mtok * Hdim * 2);
  unsigned short* shb   = (unsigned short*)alloc((size_t)Mtok * Hdim * 2);
  float*          x2f   = (float*)alloc((size_t)Mtok * Hdim * 4);
  float*          fused = x2f;   // x2 is dead after LayerNorm -> reuse
  int*   selSyn = (int*)alloc((size_t)Mtok * 4);
  int*   selLen = (int*)alloc((size_t)Mtok * 4);
  int*   selSem = (int*)alloc((size_t)Mtok * 4);
  float* gSyn   = (float*)alloc((size_t)Mtok * 4);
  float* gLen   = (float*)alloc((size_t)Mtok * 4);
  float* gSem   = (float*)alloc((size_t)Mtok * 4);

  dim3 blk(THREADS);

  // bf16 conversions
  long nHS = (long)Mtok * Hdim;
  k_cvt<<<(unsigned)((nHS + THREADS - 1) / THREADS), blk, 0, stream>>>(hs, hsb, nHS);
  unsigned gw = (unsigned)((HH + THREADS - 1) / THREADS);
  k_cvt_wt<<<gw, blk, 0, stream>>>(gcn1_w, wt[0], Hdim, Hdim);
  k_cvt_wt<<<gw, blk, 0, stream>>>(gcn2_w, wt[1], Hdim, Hdim);
  for (int e = 0; e < 3; ++e)
    k_cvt_wt<<<gw, blk, 0, stream>>>(syn_w + e * HH, wt[2 + e], Hdim, Hdim);
  k_cvt_wt<<<gw, blk, 0, stream>>>(lenS_w, wt[5], Hdim, Hdim);
  k_cvt_wt<<<gw, blk, 0, stream>>>(lenL_w, wt[6], Hdim, Hdim);
  for (int e = 0; e < 3; ++e)
    k_cvt_wt<<<gw, blk, 0, stream>>>(sem_w + e * HH, wt[7 + e], Hdim, Hdim);

  k_adjprep<<<Mtok, blk, 0, stream>>>(adj, adjnb);
  k_router<<<Mtok, blk, 0, stream>>>(hs, router_w, router_b, seqlen,
                                     selSyn, selLen, selSem, gSyn, gLen, gSem);

  dim3 gg(Hdim / BN, Mtok / BM);   // (12, 64)
  const long sBT = (long)Hdim * Sseq;

  // GCN layer 1: support1 = hs @ W1  (stored transposed per batch)
  gemm_wmma<0><<<gg, blk, 0, stream>>>(hsb, wt[0], suppT, Mtok, Hdim, Hdim,
                                       0, Sseq, nullptr, nullptr, nullptr, 0, 0);
  // x1 = relu(adjn @ support1)  -> bf16 row-major
  gemm_wmma<1><<<gg, blk, 0, stream>>>(adjnb, suppT, x1b, Mtok, Hdim, Sseq,
                                       sBT, Sseq, nullptr, nullptr, nullptr, 0, 0);
  // GCN layer 2: support2 = x1 @ W2 (transposed)
  gemm_wmma<0><<<gg, blk, 0, stream>>>(x1b, wt[1], suppT, Mtok, Hdim, Hdim,
                                       0, Sseq, nullptr, nullptr, nullptr, 0, 0);
  // x2 = relu(adjn @ support2) -> f32
  gemm_wmma<2><<<gg, blk, 0, stream>>>(adjnb, suppT, x2f, Mtok, Hdim, Sseq,
                                       sBT, Sseq, nullptr, nullptr, nullptr, 0, 0);
  // shared = LN(x2 + hs) -> bf16
  k_ln<<<Mtok, blk, 0, stream>>>(x2f, hs, ln_g, ln_b, shb);

  // experts: fused = sum of gated, per-group top-1 selected outputs
  gemm_wmma<3><<<gg, blk, 0, stream>>>(shb, wt[2], fused, Mtok, Hdim, Hdim,
                                       0, Sseq, syn_b + 0 * Hdim, selSyn, gSyn, 0, 1);
  gemm_wmma<3><<<gg, blk, 0, stream>>>(shb, wt[3], fused, Mtok, Hdim, Hdim,
                                       0, Sseq, syn_b + 1 * Hdim, selSyn, gSyn, 1, 0);
  gemm_wmma<3><<<gg, blk, 0, stream>>>(shb, wt[4], fused, Mtok, Hdim, Hdim,
                                       0, Sseq, syn_b + 2 * Hdim, selSyn, gSyn, 2, 0);
  gemm_wmma<3><<<gg, blk, 0, stream>>>(hsb, wt[5], fused, Mtok, Hdim, Hdim,
                                       0, Sseq, lenS_b, selLen, gLen, 0, 0);
  gemm_wmma<3><<<gg, blk, 0, stream>>>(hsb, wt[6], fused, Mtok, Hdim, Hdim,
                                       0, Sseq, lenL_b, selLen, gLen, 1, 0);
  gemm_wmma<3><<<gg, blk, 0, stream>>>(hsb, wt[7], fused, Mtok, Hdim, Hdim,
                                       0, Sseq, sem_b + 0 * Hdim, selSem, gSem, 0, 0);
  gemm_wmma<3><<<gg, blk, 0, stream>>>(hsb, wt[8], fused, Mtok, Hdim, Hdim,
                                       0, Sseq, sem_b + 1 * Hdim, selSem, gSem, 1, 0);
  gemm_wmma<3><<<gg, blk, 0, stream>>>(hsb, wt[9], fused, Mtok, Hdim, Hdim,
                                       0, Sseq, sem_b + 2 * Hdim, selSem, gSem, 2, 0);

  // classifier
  k_cls<<<Mtok, blk, 0, stream>>>(fused, cls_w, cls_b, out);
}
